// GConv_23046794510783
// MI455X (gfx1250) — matrix operand order repaired and checked by reading, not compiled, not run on hardware
//
#include <hip/hip_runtime.h>
#include <hip/hip_bf16.h>

// ---- types ----
typedef __attribute__((ext_vector_type(16))) __bf16          v16bf;
typedef __attribute__((ext_vector_type(8)))  float           v8f;
typedef __attribute__((ext_vector_type(16))) unsigned short  v16u;
typedef __attribute__((ext_vector_type(8)))  unsigned short  v8u;

#define D_FEAT 256

// native f32 -> bf16 (RNE); lets clang use hw cvt ops on gfx1250
__device__ __forceinline__ unsigned short f2bf(float f) {
    return __builtin_bit_cast(unsigned short, (__bf16)f);
}
__device__ __forceinline__ float bf2f(unsigned short s) {
    return __builtin_bit_cast(float, ((unsigned)s) << 16);
}

// ---- zero fill (float4 stores) ----
__global__ void gcn_zero_f32(float* __restrict__ p, long n) {
    long idx = ((long)blockIdx.x * blockDim.x + threadIdx.x) * 4;
    if (idx + 3 < n) {
        *(float4*)(p + idx) = make_float4(0.f, 0.f, 0.f, 0.f);
    } else {
        for (int j = 0; j < 4; ++j) if (idx + j < n) p[idx + j] = 0.f;
    }
}

// ---- x (f32 row-major) -> xb (bf16 row-major), 8 elems/thread ----
__global__ void gcn_cvt_x(const float* __restrict__ x, unsigned short* __restrict__ xb, long n) {
    long i = ((long)blockIdx.x * blockDim.x + threadIdx.x) * 8;
    if (i >= n) return;
    float4 a = *(const float4*)(x + i);
    float4 b = *(const float4*)(x + i + 4);
    v8u o;
    o[0] = f2bf(a.x); o[1] = f2bf(a.y); o[2] = f2bf(a.z); o[3] = f2bf(a.w);
    o[4] = f2bf(b.x); o[5] = f2bf(b.y); o[6] = f2bf(b.z); o[7] = f2bf(b.w);
    *(v8u*)(xb + i) = o;
}

// ---- degree accumulation ----
__global__ void gcn_deg(const int* __restrict__ row, const float* __restrict__ vals,
                        float* __restrict__ deg, int E) {
    int e = blockIdx.x * blockDim.x + threadIdx.x;
    if (e < E) unsafeAtomicAdd(&deg[row[e]], vals[e]);
}

// ---- dinv = deg > 0 ? deg^-1/2 : 0 ----
__global__ void gcn_dinv(const float* __restrict__ deg, float* __restrict__ dinv, int N) {
    int i = blockIdx.x * blockDim.x + threadIdx.x;
    if (i < N) {
        float d = deg[i];
        dinv[i] = d > 0.f ? rsqrtf(d) : 0.f;
    }
}

// ---- pack w [256x256 f32 row-major] into WMMA-B bf16 fragments ----
// wp[nt][ks][lane] = 16 contiguous bf16 (32B) per lane.
// B fragment (32x16, KxN), wave32: lanes 0-15 -> N=lane,   K=kb+0..7,16..23 (2/VGPR)
//                                  lanes 16-31 -> N=lane-16, K=kb+8..15,24..31
__global__ void gcn_packw(const float* __restrict__ w, unsigned short* __restrict__ wp) {
    int t = blockIdx.x * blockDim.x + threadIdx.x;   // 0..4095 = 16 nt * 8 ks * 32 lanes
    int lane = t & 31;
    int ks   = (t >> 5) & 7;
    int nt   = t >> 8;
    int ncol = nt * 16 + (lane & 15);
    int kb   = ks * 32 + ((lane >> 4) ? 8 : 0);
    v16u o;
#pragma unroll
    for (int j = 0; j < 16; ++j)
        o[j] = f2bf(w[(kb + (j < 8 ? j : j + 8)) * D_FEAT + ncol]);
    *((v16u*)wp + t) = o;
}

// ---- GEMM h = bf16(xb @ w), WMMA 16x16x32 bf16, f32 accumulate ----
// block = (32, 8): 8 waves; wave owns one 16x16 output tile.
// grid = (N/16, 2): blockIdx.y*8+threadIdx.y = n-tile.
// Inner loop is pure loads + wmma: A slice is contiguous bf16 in xb.
__global__ void gcn_gemm(const unsigned short* __restrict__ xb,
                         const unsigned short* __restrict__ wp,
                         unsigned short* __restrict__ h, int N) {
    int lane = threadIdx.x;
    int nt   = blockIdx.y * 8 + threadIdx.y;
    int m0   = blockIdx.x * 16;
    int mi   = lane & 15;
    int koff = (lane >> 4) * 8;      // A layout: lanes 0-15 K=0..7/16..23; 16-31 K=8..15/24..31
    int row  = m0 + mi; if (row >= N) row = N - 1;
    const unsigned short* xr = xb + (long)row * D_FEAT;
    const v16u* bp = (const v16u*)wp + (nt * 8) * 32 + lane;

    v8f acc = {0.f, 0.f, 0.f, 0.f, 0.f, 0.f, 0.f, 0.f};
#pragma unroll
    for (int ks = 0; ks < 8; ++ks) {
        int kb = ks * 32;
        v8u alo = *(const v8u*)(xr + kb + koff);        // K = kb+koff+0..7
        v8u ahi = *(const v8u*)(xr + kb + 16 + koff);   // K = kb+16+koff+0..7
        v16u au = __builtin_shufflevector(alo, ahi,
                    0, 1, 2, 3, 4, 5, 6, 7, 8, 9, 10, 11, 12, 13, 14, 15);
        v16bf a = __builtin_bit_cast(v16bf, au);
        v16bf b = __builtin_bit_cast(v16bf, bp[ks * 32]);
        acc = __builtin_amdgcn_wmma_f32_16x16x32_bf16(
            /*neg_a=*/false, a, /*neg_b=*/false, b,
            /*c_mod=*/(short)0, acc, /*reuse_a=*/false, /*reuse_b=*/false);
    }

    // C layout: VGPR r -> lanes 0-15: M=r, lanes 16-31: M=8+r; N = lane&15.
    int colv = nt * 16 + (lane & 15);
    int rb   = m0 + ((lane >> 4) * 8);
#pragma unroll
    for (int r = 0; r < 8; ++r) {
        int rr = rb + r;
        if (rr < N) h[(long)rr * D_FEAT + colv] = f2bf(acc[r]);
    }
}

// ---- edge aggregation: one wave per edge, lane handles 8 features ----
__global__ void gcn_agg(const int* __restrict__ row, const int* __restrict__ col,
                        const float* __restrict__ vals, const float* __restrict__ dinv,
                        const unsigned short* __restrict__ h, float* __restrict__ out, int E) {
    long t = (long)blockIdx.x * blockDim.x + threadIdx.x;
    int e = (int)(t >> 5);
    int lane = threadIdx.x & 31;
    if (e >= E) return;
    int r = row[e], c = col[e];
    float nv = vals[e] * dinv[r] * dinv[c];
    v8u hv = *(const v8u*)(h + (long)c * D_FEAT + lane * 8);
    float* op = out + (long)r * D_FEAT + lane * 8;
#pragma unroll
    for (int j = 0; j < 8; ++j)
        unsafeAtomicAdd(op + j, bf2f(hv[j]) * nv);
}

// ---- bias + relu (in place on d_out) ----
__global__ void gcn_bias_relu(float* __restrict__ out, const float* __restrict__ b, long n) {
    long i = (long)blockIdx.x * blockDim.x + threadIdx.x;
    if (i < n) {
        float v = out[i] + b[i & (D_FEAT - 1)];
        out[i] = v > 0.f ? v : 0.f;
    }
}

extern "C" void kernel_launch(void* const* d_in, const int* in_sizes, int n_in,
                              void* d_out, int out_size, void* d_ws, size_t ws_size,
                              hipStream_t stream) {
    const float* x     = (const float*)d_in[0];
    const int*   erow  = (const int*)d_in[1];
    const int*   ecol  = (const int*)d_in[2];
    const float* evals = (const float*)d_in[3];
    const float* w     = (const float*)d_in[4];
    const float* bias  = (const float*)d_in[5];
    float* out = (float*)d_out;

    int N = in_sizes[0] / D_FEAT;   // 100000
    int E = in_sizes[1];            // 3200000

    char* ws = (char*)d_ws;
    float*          deg  = (float*)(ws);                        // N f32    (400 KB)
    float*          dinv = (float*)(ws + (1l << 20));           // N f32    (400 KB)
    unsigned short* wp   = (unsigned short*)(ws + (2l << 20));  // 128 KB packed bf16 W
    unsigned short* xb   = (unsigned short*)(ws + (4l << 20));  // N*256 bf16 (51.2 MB)
    unsigned short* h    = (unsigned short*)(ws + (60l << 20)); // N*256 bf16 (51.2 MB)

    long outN = (long)N * D_FEAT;

    // 1) zero accumulators (required every call: graph replays reuse buffers)
    gcn_zero_f32<<<(int)((outN / 4 + 255) / 256), 256, 0, stream>>>(out, outN);
    gcn_zero_f32<<<(N + 1023) / 1024, 256, 0, stream>>>(deg, (long)N);

    // 2) x -> bf16 (one bandwidth pass; removes all cvt VALU from the GEMM loop)
    gcn_cvt_x<<<(int)((outN / 8 + 255) / 256), 256, 0, stream>>>(x, xb, outN);

    // 3) degree
    gcn_deg<<<(E + 255) / 256, 256, 0, stream>>>(erow, evals, deg, E);

    // 4) dinv
    gcn_dinv<<<(N + 255) / 256, 256, 0, stream>>>(deg, dinv, N);

    // 5) pack weights into WMMA-B fragment order
    gcn_packw<<<16, 256, 0, stream>>>(w, wp);

    // 6) WMMA GEMM: h = bf16(xb @ w)
    dim3 gg((N + 15) / 16, 2), gb(32, 8);
    gcn_gemm<<<gg, gb, 0, stream>>>(xb, wp, h, N);

    // 7) edge gather/scale/scatter-add (L2-resident: h 51MB + out 102MB < 192MB L2)
    long aggThreads = (long)E * 32;
    gcn_agg<<<(int)((aggThreads + 255) / 256), 256, 0, stream>>>(erow, ecol, evals, dinv, h, out, E);

    // 8) bias + relu
    gcn_bias_relu<<<(int)((outN + 255) / 256), 256, 0, stream>>>(out, bias, outN);
}
